// LightRNNDecoder_60026462929278
// MI455X (gfx1250) — matrix operand order repaired and checked by reading, not compiled, not run on hardware
//
#include <hip/hip_runtime.h>

#define NTOK 2048
#define DEMB 768
#define TS   256
#define LDA  776   // A staging: 768 + 8 halves pad (16B align kept, bank conflicts broken)
#define LDC  257
#define KCB  32    // K rows per async B chunk (= one WMMA k-step)
#define BSTR 36    // f32 row stride in async B staging (144B, 16B aligned)
#define NCHUNK (DEMB / KCB)   // 24

typedef __attribute__((ext_vector_type(16))) _Float16 v16h;
typedef __attribute__((ext_vector_type(8)))  _Float16 v8h;
typedef __attribute__((ext_vector_type(8)))  float    v8f;

// ---- WMMA fragment helpers (wave32; A and B both use contiguous k = 16*(lane/16) + s) ----

// A: 16x32 f16 from LDS. lanes 0-15: M=0..15, K=0..15; lanes 16-31: K=16..31.
__device__ __forceinline__ v16h load_a_frag(const _Float16* ldsA, int lane, int k0) {
  const int row = lane & 15;
  const int kb  = (lane >> 4) << 4;
  const _Float16* p = ldsA + row * LDA + k0 + kb;
  v8h lo = *(const v8h*)(p);
  v8h hi = *(const v8h*)(p + 8);
  v16h a;
#pragma unroll
  for (int e = 0; e < 8; ++e) { a[e] = lo[e]; a[e + 8] = hi[e]; }
  return a;
}

// B: 32x16, lane = column n, lanes 0-15 hold K=0..15, lanes 16-31 K=16..31.
// W_row is (256 x 768) row-major: column n of B is contiguous in memory (k fast).
__device__ __forceinline__ v16h load_b_rowmajor_n(const float* W, int lane, int n0, int k0) {
  const int n  = n0 + (lane & 15);
  const int kb = (lane >> 4) << 4;
  const float* p = W + (size_t)n * DEMB + k0 + kb;
  v16h b;
#pragma unroll
  for (int e = 0; e < 16; ++e) b[e] = (_Float16)p[e];
  return b;
}

// B fragment from raw-f32 LDS chunk (k-major, row stride BSTR floats), convert on the fly.
__device__ __forceinline__ v16h load_b_f32lds(const float* bwf, int lane, int nloc0) {
  const int n  = nloc0 + (lane & 15);
  const int kb = (lane >> 4) << 4;
  const float* p = bwf + kb * BSTR + n;
  v16h b;
#pragma unroll
  for (int e = 0; e < 16; ++e) b[e] = (_Float16)p[e * BSTR];
  return b;
}

// C/D: lane 0-15 -> n = lane, rows 0..7 in VGPR 0..7 ; lanes 16-31 -> rows 8..15.
__device__ __forceinline__ void store_c_bias(float* ldsL, int lane, int n0, v8f c, const float* bias) {
  const int n  = n0 + (lane & 15);
  const int mb = (lane >> 4) << 3;
  const float bv = bias[n];
#pragma unroll
  for (int j = 0; j < 8; ++j) ldsL[(mb + j) * LDC + n] = c[j] + bv;
}

__device__ __forceinline__ float nll_from_lds(const float* ldsL, int m, int s, int target) {
  float mx = -3.0e38f;
#pragma unroll
  for (int i = 0; i < 16; ++i) mx = fmaxf(mx, ldsL[m * LDC + s + 16 * i]);
#pragma unroll
  for (int off = 8; off > 0; off >>= 1) mx = fmaxf(mx, __shfl_xor(mx, off, 16));
  float se = 0.0f;
#pragma unroll
  for (int i = 0; i < 16; ++i) se += __expf(ldsL[m * LDC + s + 16 * i] - mx);
#pragma unroll
  for (int off = 8; off > 0; off >>= 1) se += __shfl_xor(se, off, 16);
  return -(ldsL[m * LDC + target] - mx - __logf(se));  // valid on s == 0
}

// ---- async copy helpers (inline asm: bypasses builtin pointer-AS typing) ----
__device__ __forceinline__ unsigned lds_off_u32(void* p) {
  // generic -> AS(3) addrspacecast; AS(3) pointers are 32-bit wave-relative LDS offsets
  return (unsigned)(unsigned long long)(__attribute__((address_space(3))) void*)p;
}

__device__ __forceinline__ void async_ld_b128(void* lds_dst, const void* gsrc) {
  asm volatile("global_load_async_to_lds_b128 %0, %1, off"
               :: "v"(lds_off_u32(lds_dst)), "v"((unsigned long long)gsrc)
               : "memory");
}

__device__ __forceinline__ void wait_async_le(int n) {
#if __has_builtin(__builtin_amdgcn_s_wait_asynccnt)
  if (n == 0) __builtin_amdgcn_s_wait_asynccnt(0);
  else        __builtin_amdgcn_s_wait_asynccnt(8);
#else
  if (n == 0) asm volatile("s_wait_asynccnt 0" ::: "memory");
  else        asm volatile("s_wait_asynccnt 8" ::: "memory");
#endif
}

// Issue one 32(K)x32(N) f32 chunk of col_weight[r] into this wave's LDS buffer.
// 8 x global_load_async_to_lds_b128: 4 rows per instr, lane = (row in quad, 16B segment);
// each 128B cacheline (one wave's 32-col window of a K-row) is fetched exactly once.
__device__ __forceinline__ void issue_b_chunk(const float* Wr, int k0, int n0a,
                                              float* bwf, int lane) {
  const int rrow = lane >> 3;   // 0..3
  const int seg  = lane & 7;    // 0..7
#pragma unroll
  for (int i = 0; i < 8; ++i) {
    const int kl = i * 4 + rrow;
    const float* g = Wr + (size_t)(k0 + kl) * TS + n0a + seg * 4;
    float* l = bwf + kl * BSTR + seg * 4;
    async_ld_b128(l, g);
  }
}

// ---------------- kernel 0: init accumulators ----------------
__global__ void init_kernel(float* sums) { sums[0] = 0.0f; sums[1] = 0.0f; }

// ---------------- kernel 1: row head (GEMM + fused CE) ----------------
__global__ __launch_bounds__(256) void row_head_kernel(
    const float* __restrict__ hs, const long long* __restrict__ tids,
    const float* __restrict__ Wrow, const float* __restrict__ brow,
    float* __restrict__ row_sum) {
  __shared__ __align__(16) _Float16 ldsA[16 * LDA];
  __shared__ float ldsL[16 * LDC];
  const int t = threadIdx.x;
  const int tile0 = blockIdx.x * 16;

  for (int idx = t; idx < 16 * DEMB; idx += 256) {
    const int m = idx / DEMB, k = idx % DEMB;
    ldsA[m * LDA + k] = (_Float16)hs[(size_t)(tile0 + m) * DEMB + k];
  }
  __syncthreads();

  const int lane = t & 31, wave = t >> 5;
  const int n0a = wave * 32, n0b = n0a + 16;
  v8f c0 = {}, c1 = {};
#pragma unroll 4
  for (int k0 = 0; k0 < DEMB; k0 += 32) {
    v16h a  = load_a_frag(ldsA, lane, k0);
    v16h b0 = load_b_rowmajor_n(Wrow, lane, n0a, k0);
    v16h b1 = load_b_rowmajor_n(Wrow, lane, n0b, k0);
    c0 = __builtin_amdgcn_wmma_f32_16x16x32_f16(false, a, false, b0, (short)0, c0, false, false);
    c1 = __builtin_amdgcn_wmma_f32_16x16x32_f16(false, a, false, b1, (short)0, c1, false, false);
  }
  store_c_bias(ldsL, lane, n0a, c0, brow);
  store_c_bias(ldsL, lane, n0b, c1, brow);
  __syncthreads();

  const int m = wave * 2 + (lane >> 4);
  const int s = lane & 15;
  const long long id = tids[tile0 + m];
  const int rid = (int)(id >> 8) & 255;
  const float loss = nll_from_lds(ldsL, m, s, rid);
  if (s == 0) atomicAdd(row_sum, loss);
}

// ---------------- kernel 2: bucket tokens by row id (single WG) ----------------
__global__ __launch_bounds__(256) void bucket_kernel(
    const long long* __restrict__ tids, int* __restrict__ offsets, int* __restrict__ bucket) {
  __shared__ int cnt[TS];
  __shared__ int cur[TS];
  const int t = threadIdx.x;
  cnt[t] = 0;
  __syncthreads();
  for (int i = t; i < NTOK; i += 256) atomicAdd(&cnt[(int)(tids[i] >> 8) & 255], 1);
  __syncthreads();
  if (t == 0) {
    int run = 0;
    for (int r = 0; r < TS; ++r) { cur[r] = run; offsets[r] = run; run += cnt[r]; }
    offsets[TS] = run;
  }
  __syncthreads();
  for (int i = t; i < NTOK; i += 256) {
    const int rid = (int)(tids[i] >> 8) & 255;
    bucket[atomicAdd(&cur[rid], 1)] = i;
  }
}

// ---------------- kernel 3: col head (grouped GEMM + fused CE), one WG per row code ----------------
// 201 MB col_weight stream: per-wave double-buffered ASYNC Global->LDS copies (ASYNCcnt),
// WMMA consumes chunk c while chunk c+1 is in flight. No VGPR staging round-trip.
__global__ __launch_bounds__(256) void col_head_kernel(
    const float* __restrict__ hs, const long long* __restrict__ tids,
    const float* __restrict__ Wcol, const float* __restrict__ cbias,
    const int* __restrict__ offsets, const int* __restrict__ bucket,
    float* __restrict__ col_sum) {
  __shared__ __align__(16) _Float16 ldsA[16 * LDA];
  __shared__ __align__(16) float ldsBF[8 * 2 * KCB * BSTR];   // 8 waves x 2 phases x 4.5KB
  __shared__ float ldsL[16 * LDC];
  __shared__ int toks[16];
  const int r = blockIdx.x;
  const int t = threadIdx.x;
  const int base = offsets[r];
  const int cnt  = offsets[r + 1] - base;
  if (cnt == 0) return;  // block-uniform

  const float* Wr = Wcol + (size_t)r * DEMB * TS;
  const float* br = cbias + (size_t)r * TS;
  const int lane = t & 31, wave = t >> 5;
  const int n0a = wave * 32, n0b = n0a + 16;
  const int ntiles = (cnt + 15) >> 4;

  for (int mt = 0; mt < ntiles; ++mt) {
    if (t < 16) toks[t] = (mt * 16 + t < cnt) ? bucket[base + mt * 16 + t] : -1;
    __syncthreads();
    for (int idx = t; idx < 16 * DEMB; idx += 256) {
      const int m = idx / DEMB, k = idx % DEMB;
      const int tok = toks[m];
      ldsA[m * LDA + k] = (tok >= 0) ? (_Float16)hs[(size_t)tok * DEMB + k] : (_Float16)0.0f;
    }
    __syncthreads();

    v8f c0 = {}, c1 = {};
    float* bw0 = ldsBF + wave * 2 * KCB * BSTR;
    float* bw1 = bw0 + KCB * BSTR;
    issue_b_chunk(Wr, 0, n0a, bw0, lane);
    for (int c = 0; c < NCHUNK; ++c) {
      float* cur = (c & 1) ? bw1 : bw0;
      if (c + 1 < NCHUNK) {
        issue_b_chunk(Wr, (c + 1) * KCB, n0a, ((c + 1) & 1) ? bw1 : bw0, lane);
        wait_async_le(8);   // chunk c complete (in-order), chunk c+1 may be in flight
      } else {
        wait_async_le(0);
      }
      v16h a  = load_a_frag(ldsA, lane, c * KCB);
      v16h b0 = load_b_f32lds(cur, lane, 0);
      v16h b1 = load_b_f32lds(cur, lane, 16);
      c0 = __builtin_amdgcn_wmma_f32_16x16x32_f16(false, a, false, b0, (short)0, c0, false, false);
      c1 = __builtin_amdgcn_wmma_f32_16x16x32_f16(false, a, false, b1, (short)0, c1, false, false);
    }
    store_c_bias(ldsL, lane, n0a, c0, br);
    store_c_bias(ldsL, lane, n0b, c1, br);
    __syncthreads();

    const int m = wave * 2 + (lane >> 4);
    const int s = lane & 15;
    const int tok = toks[m];
    if (tok >= 0) {
      const int cid = (int)(tids[tok]) & 255;
      const float loss = nll_from_lds(ldsL, m, s, cid);
      if (s == 0) atomicAdd(col_sum, loss);
    }
    __syncthreads();  // protect toks/ldsA for next tile
  }
}

// ---------------- kernel 4: finalize ----------------
__global__ void finalize_kernel(const float* sums, float* out) {
  out[0] = (sums[0] + sums[1]) * (1.0f / (float)NTOK);
}

extern "C" void kernel_launch(void* const* d_in, const int* in_sizes, int n_in,
                              void* d_out, int out_size, void* d_ws, size_t ws_size,
                              hipStream_t stream) {
  const float*     hs    = (const float*)d_in[0];
  const long long* tids  = (const long long*)d_in[1];
  const float*     Wrow  = (const float*)d_in[2];
  const float*     brow  = (const float*)d_in[3];
  const float*     Wcol  = (const float*)d_in[4];
  const float*     cbias = (const float*)d_in[5];
  float* out = (float*)d_out;

  float* sums    = (float*)d_ws;                       // [0]=row loss, [1]=col loss
  int*   offsets = (int*)((char*)d_ws + 16);           // 257 ints
  int*   bucket  = offsets + 260;                      // 2048 ints

  init_kernel<<<1, 1, 0, stream>>>(sums);
  row_head_kernel<<<NTOK / 16, 256, 0, stream>>>(hs, tids, Wrow, brow, sums + 0);
  bucket_kernel<<<1, 256, 0, stream>>>(tids, offsets, bucket);
  col_head_kernel<<<TS, 256, 0, stream>>>(hs, tids, Wcol, cbias, offsets, bucket, sums + 1);
  finalize_kernel<<<1, 1, 0, stream>>>(sums, out);
}
